// ClusterMemory_part_55456617726497
// MI455X (gfx1250) — matrix operand order repaired and checked by reading, not compiled, not run on hardware
//
#include <hip/hip_runtime.h>
#include <hip/hip_bf16.h>
#include <stdint.h>

// ---------------------------------------------------------------------------
// CDNA5 / gfx1250 fused kernel for the ClusterMemory reference.
//   banks are read ONCE: TDM streams each 64x32 fp32 tile into LDS and also
//   DMA-copies it to nf (tensor_store_from_lds, ordered after the load);
//   waves only convert fp32->bf16 and run v_wmma_f32_16x16x32_bf16.
//   CE uses online sum-exp (|logit| <= 20, so no max pass needed).
// ---------------------------------------------------------------------------

typedef __attribute__((ext_vector_type(16))) __bf16 v16bf;
typedef __attribute__((ext_vector_type(8)))  __bf16 v8bf;
typedef __attribute__((ext_vector_type(8)))  float  v8f;
typedef __attribute__((ext_vector_type(4)))  unsigned int u32x4;
typedef __attribute__((ext_vector_type(8)))  int i32x8;
typedef __attribute__((ext_vector_type(4)))  int i32x4;

#define NFEAT 2048
#define NSAMP 32768
#define BATCH 256
#define INV_TEMP 20.0f        // 1 / 0.05
#define KCHUNKS (NFEAT / 32)  // 64 chunks of K=32 (bf16 WMMA depth)

__device__ __forceinline__ float blockReduceSum(float v, float* red) {
  const int tid = threadIdx.x;
  red[tid] = v;
  __syncthreads();
  #pragma unroll
  for (int off = 128; off > 0; off >>= 1) {
    if (tid < off) red[tid] += red[tid + off];
    __syncthreads();
  }
  float r = red[0];
  __syncthreads();
  return r;
}

// ---- TDM: issue one (global->LDS load, LDS->global store) descriptor pair -
// D# per CDNA5 ISA ch.8: 2-D tile 32 (dim0, contiguous) x 64 rows of f32,
// row stride 2048 elements. TDM ops from one wave execute in order, so the
// store (the features->nf copy) reads LDS only after the load filled it.
// clang-23 toolchain: 6-arg builtin (g0, g1, g2, g3, extra, cpol).
__device__ __forceinline__ void tdm_issue_pair(const float* gsrc, float* gdst,
                                               unsigned lds_off) {
  const unsigned long long ga = (unsigned long long)(uintptr_t)gsrc;
  const unsigned long long gd = (unsigned long long)(uintptr_t)gdst;
  // group0: [1:0]=count=1 | [63:32]=lds_addr | [120:64]=global_addr |
  //         [127:126]=type=2
  u32x4 g0l = {1u, lds_off, (unsigned)(ga & 0xffffffffu),
               (unsigned)((ga >> 32) & 0x01ffffffu) | 0x80000000u};
  u32x4 g0s = {1u, lds_off, (unsigned)(gd & 0xffffffffu),
               (unsigned)((gd >> 32) & 0x01ffffffu) | 0x80000000u};
  // group1: data_size=2 (4B) @ [17:16]; tensor_dim0=2048 @ [79:48];
  //         tensor_dim1=32768 @ [111:80]; tile_dim0=32 @ [127:112];
  //         tile_dim1=64 @ [143:128]; tensor_dim0_stride=2048 @ [207:160]
  i32x8 g1 = {0x00020000,       // workgroup_mask=0, data_size=4B
              0x08000000,       // tensor_dim0(2048) low16 -> [31:16]
              (int)0x80000000,  // tensor_dim1(32768) low16 -> [31:16]
              0x00200000,       // tile_dim0(32) -> [31:16]
              64,               // tile_dim1(64) -> [15:0]
              2048,             // tensor_dim0_stride[31:0]
              0, 0};
  i32x4 z4 = {0, 0, 0, 0};      // 2-D tensor: groups 2/3 unused
  i32x8 z8 = {0, 0, 0, 0, 0, 0, 0, 0};
  __builtin_amdgcn_tensor_load_to_lds(g0l, g1, z4, z4, z8, 0);
  __builtin_amdgcn_tensor_store_from_lds(g0s, g1, z4, z4, z8, 0);
}

// ---- kernel 0: zero the workspace accumulators (sumexp, tlogit, ld) -------
__global__ void k_init(float* __restrict__ p, int n) {
  for (int i = threadIdx.x + blockIdx.x * blockDim.x; i < n;
       i += blockDim.x * gridDim.x)
    p[i] = 0.f;
}

// ---- kernel 1: row-normalize student+teacher, write xn(f32) & x(bf16), ----
// ---- and accumulate ld = mean_B sum_F (x - t)^2 per view -------------------
__global__ __launch_bounds__(256) void k_norm(
    const float* __restrict__ s0, const float* __restrict__ s1,
    const float* __restrict__ s2, const float* __restrict__ t0,
    const float* __restrict__ t1, const float* __restrict__ t2,
    float* __restrict__ xn, unsigned short* __restrict__ xbf_raw,
    float* __restrict__ ld_acc) {
  __shared__ float red[256];
  const int v = blockIdx.y, row = blockIdx.x, tid = threadIdx.x;
  const float* S = (v == 0 ? s0 : v == 1 ? s1 : s2) + (size_t)row * NFEAT;
  const float* T = (v == 0 ? t0 : v == 1 ? t1 : t2) + (size_t)row * NFEAT;
  float* XN = xn + ((size_t)v * BATCH + row) * NFEAT;
  __bf16* XB = (__bf16*)xbf_raw + ((size_t)v * BATCH + row) * NFEAT;

  const float4* Sp = (const float4*)S;
  const float4* Tp = (const float4*)T;
  float4 a0 = Sp[tid * 2], a1 = Sp[tid * 2 + 1];
  float4 b0 = Tp[tid * 2], b1 = Tp[tid * 2 + 1];
  float xs[8] = {a0.x, a0.y, a0.z, a0.w, a1.x, a1.y, a1.z, a1.w};
  float ts[8] = {b0.x, b0.y, b0.z, b0.w, b1.x, b1.y, b1.z, b1.w};

  float ssx = 0.f, sst = 0.f;
  #pragma unroll
  for (int e = 0; e < 8; ++e) { ssx += xs[e] * xs[e]; sst += ts[e] * ts[e]; }
  const float nx = blockReduceSum(ssx, red);
  const float nt = blockReduceSum(sst, red);
  const float rx = rsqrtf(nx), rt = rsqrtf(nt);

  float xv[8];
  float d2 = 0.f;
  #pragma unroll
  for (int e = 0; e < 8; ++e) {
    xv[e] = xs[e] * rx;
    const float d = xv[e] - ts[e] * rt;
    d2 += d * d;
  }
  float4* XNp = (float4*)XN;
  XNp[tid * 2]     = make_float4(xv[0], xv[1], xv[2], xv[3]);
  XNp[tid * 2 + 1] = make_float4(xv[4], xv[5], xv[6], xv[7]);
  #pragma unroll
  for (int e = 0; e < 8; ++e) XB[tid * 8 + e] = (__bf16)xv[e];

  const float td = blockReduceSum(d2, red);
  if (tid == 0) atomicAdd(ld_acc + v, td * (1.0f / BATCH));
}

// ---- kernel 2: fused GEMM (bf16 WMMA) + TDM bank->nf copy + CE stats ------
// grid = (512, 3): blockIdx.y = bank, each WG owns 64 bank rows x all 256
// batch rows. 8 waves/WG, each wave: 2 M-subtiles x 4 N-subtiles of 16x16.
// Double-buffered TDM pipeline: wave 0 issues chunk k+1 while all waves
// convert+multiply chunk k.
__global__ __launch_bounds__(256) void k_gemm(
    const float* __restrict__ f0, const float* __restrict__ f1,
    const float* __restrict__ f2, const unsigned short* __restrict__ xbf_raw,
    const int* __restrict__ targets, float* __restrict__ nf,
    float* __restrict__ sumexp, float* __restrict__ tlogit) {
  __shared__ float Braw[2][64 * 32];  // TDM-written raw fp32 chunks (16 KB)
  __shared__ __bf16 Bbf[2][64][40];   // bf16 staging, padded stride (10 KB)
  __shared__ int tgt[BATCH];

  const int v = blockIdx.y;
  const float* F = (v == 0 ? f0 : v == 1 ? f1 : f2);
  const __bf16* X = (const __bf16*)xbf_raw + (size_t)v * BATCH * NFEAT;
  float* NF = nf + (size_t)v * NSAMP * NFEAT;
  float* SE = sumexp + v * BATCH;
  float* TL = tlogit + v * BATCH;

  const int tid = threadIdx.x;
  const size_t j0 = (size_t)blockIdx.x * 64;

  tgt[tid] = targets[tid];

  const int w = tid >> 5, lane = tid & 31, half = lane >> 4, lr = lane & 15;
  const int cr = tid >> 2, cq = tid & 3;  // convert map: 8 floats per thread

  const unsigned lds0 = (unsigned)(unsigned long long)(uintptr_t)&Braw[0][0];
  const unsigned lds1 = (unsigned)(unsigned long long)(uintptr_t)&Braw[1][0];

  v8f acc[2][4] = {};

  if (w == 0)  // TDM is per-wave; one issuing wave per WG
    tdm_issue_pair(F + j0 * NFEAT, NF + j0 * NFEAT, lds0);

  int buf = 0;
  for (int kc = 0; kc < KCHUNKS; ++kc) {
    // in-order completion: <=1 outstanding leaves only the paired store,
    // so the load of chunk kc has landed in Braw[buf].
    if (w == 0) __builtin_amdgcn_s_wait_tensorcnt(1);
    __syncthreads();  // propagate arrival to all 8 waves

    if (w == 0 && kc + 1 < KCHUNKS)
      tdm_issue_pair(F + j0 * NFEAT + (kc + 1) * 32,
                     NF + j0 * NFEAT + (kc + 1) * 32, buf ? lds0 : lds1);

    // fp32 -> bf16 conversion into the WMMA staging buffer
    {
      const float* src = &Braw[buf][cr * 32 + cq * 8];
      __bf16* dst = &Bbf[buf][cr][cq * 8];
      #pragma unroll
      for (int e = 0; e < 8; ++e) dst[e] = (__bf16)src[e];
    }
    __syncthreads();  // bf16 tile ready

    // A fragments direct from L2-resident bf16 X (rows are wave-private).
    // ISA 16-bit A 16x32 layout: lane = M(0..15), half selects
    // K in {h*8..h*8+7} U {16+h*8..16+h*8+7}.
    v16bf a[2], b[4];
    #pragma unroll
    for (int mi = 0; mi < 2; ++mi) {
      const __bf16* Ar =
          X + (size_t)((2 * w + mi) * 16 + lr) * NFEAT + kc * 32;
      v8bf lo = *(const v8bf*)(Ar + half * 8);
      v8bf hi = *(const v8bf*)(Ar + 16 + half * 8);
      a[mi] = __builtin_shufflevector(lo, hi, 0, 1, 2, 3, 4, 5, 6, 7, 8, 9,
                                      10, 11, 12, 13, 14, 15);
      if (kc + 1 < KCHUNKS)
        __builtin_prefetch((const void*)(Ar + 32), 0, 0);
    }
    // B fragments from LDS. B(32x16): lane&15 = N, half selects
    // K in {h*8..h*8+7} U {16+h*8..16+h*8+7} (same order as A).
    #pragma unroll
    for (int ni = 0; ni < 4; ++ni) {
      const __bf16* Br = &Bbf[buf][ni * 16 + lr][0];
      v8bf lo = *(const v8bf*)(Br + half * 8);
      v8bf hi = *(const v8bf*)(Br + 16 + half * 8);
      b[ni] = __builtin_shufflevector(lo, hi, 0, 1, 2, 3, 4, 5, 6, 7, 8, 9,
                                      10, 11, 12, 13, 14, 15);
    }
    #pragma unroll
    for (int mi = 0; mi < 2; ++mi)
      #pragma unroll
      for (int ni = 0; ni < 4; ++ni)
        acc[mi][ni] = __builtin_amdgcn_wmma_f32_16x16x32_bf16(
            false, a[mi], false, b[ni], (short)0, acc[mi][ni], false, false);

    buf ^= 1;
  }
  if (w == 0) __builtin_amdgcn_s_wait_tensorcnt(0);  // drain final DMA store

  // epilogue: C layout -> VGPR e holds M = e + 8*half, lane&15 = N.
  // sum exp(logit) over this WG's 64 columns, grab target logits.
  #pragma unroll
  for (int mi = 0; mi < 2; ++mi) {
    #pragma unroll
    for (int e = 0; e < 8; ++e) {
      const int m = (2 * w + mi) * 16 + e + 8 * half;
      float s = 0.f;
      #pragma unroll
      for (int ni = 0; ni < 4; ++ni) {
        const float val = acc[mi][ni][e] * INV_TEMP;
        s += __expf(val);
        const int n = (int)j0 + ni * 16 + lr;
        if (tgt[m] == n) TL[m] = val;  // unique writer across the grid
      }
      // reduce over the 16 lanes sharing m (xor masks stay within half-wave)
      s += __shfl_xor(s, 1);
      s += __shfl_xor(s, 2);
      s += __shfl_xor(s, 4);
      s += __shfl_xor(s, 8);
      if (lr == 0) atomicAdd(&SE[m], s);
    }
  }
}

// ---- kernel 3: finalize loss scalar ---------------------------------------
__global__ __launch_bounds__(256) void k_loss(
    const float* __restrict__ sumexp, const float* __restrict__ tlogit,
    const float* __restrict__ ld, float* __restrict__ out) {
  __shared__ float red[256];
  const int tid = threadIdx.x;
  float total = 0.f;
  for (int v = 0; v < 3; ++v) {
    const float ce = logf(sumexp[v * BATCH + tid]) - tlogit[v * BATCH + tid];
    const float cesum = blockReduceSum(ce, red);
    if (tid == 0) total += 0.5f * (cesum * (1.0f / BATCH) + ld[v]);
  }
  if (tid == 0) out[0] = total;
}

// ---- kernel 4: sequential EMA scatter-update (one block per bank) ---------
__global__ __launch_bounds__(256) void k_ema(const int* __restrict__ targets,
                                             const int* __restrict__ epoch,
                                             const float* __restrict__ xn,
                                             float* __restrict__ nf) {
  __shared__ float red[256];
  __shared__ float rn_s;
  const int v = blockIdx.x;
  const float* X = xn + (size_t)v * BATCH * NFEAT;
  float* NF = nf + (size_t)v * NSAMP * NFEAT;
  const int tid = threadIdx.x;
  const float m = (epoch[0] == -1) ? 1.0f : 0.2f;
  const float om = 1.0f - m;
  for (int s = 0; s < BATCH; ++s) {  // strict in-order semantics for dup ids
    const int y = targets[s];
    float* rp = NF + (size_t)y * NFEAT + tid * 8;
    const float* xp = X + (size_t)s * NFEAT + tid * 8;
    float rr[8];
    float ss = 0.f;
    #pragma unroll
    for (int e = 0; e < 8; ++e) {
      rr[e] = m * rp[e] + om * xp[e];
      ss += rr[e] * rr[e];
    }
    red[tid] = ss;
    __syncthreads();
    #pragma unroll
    for (int off = 128; off > 0; off >>= 1) {
      if (tid < off) red[tid] += red[tid + off];
      __syncthreads();
    }
    if (tid == 0) rn_s = rsqrtf(red[0]);
    __syncthreads();
    const float rn = rn_s;
    #pragma unroll
    for (int e = 0; e < 8; ++e) rp[e] = rr[e] * rn;
    __syncthreads();
  }
}

// ---------------------------------------------------------------------------
extern "C" void kernel_launch(void* const* d_in, const int* in_sizes, int n_in,
                              void* d_out, int out_size, void* d_ws,
                              size_t ws_size, hipStream_t stream) {
  const float* inp  = (const float*)d_in[0];
  const float* inpu = (const float*)d_in[1];
  const float* inpd = (const float*)d_in[2];
  const float* tin  = (const float*)d_in[3];
  const float* tinu = (const float*)d_in[4];
  const float* tind = (const float*)d_in[5];
  const int* targets = (const int*)d_in[6];
  const int* epoch   = (const int*)d_in[7];
  const float* f0 = (const float*)d_in[8];
  const float* f1 = (const float*)d_in[9];
  const float* f2 = (const float*)d_in[10];

  float* out = (float*)d_out;
  float* nf = out + 1;  // [3][32768][2048] updated banks, flat in order

  // workspace layout (~9.4 MB):
  float* ws_xn = (float*)d_ws;                              // 3*256*2048 f32
  unsigned short* ws_xbf =
      (unsigned short*)(ws_xn + 3 * BATCH * NFEAT);         // 3*256*2048 bf16
  float* ws_sum = (float*)(ws_xbf + 3 * BATCH * NFEAT);     // 3*256
  float* ws_tlg = ws_sum + 3 * BATCH;                       // 3*256
  float* ws_ld  = ws_tlg + 3 * BATCH;                       // 3

  // accumulators must be re-zeroed every call (graph replays keep ws dirty)
  k_init<<<1, 256, 0, stream>>>(ws_sum, 3 * BATCH * 2 + 3);
  k_norm<<<dim3(BATCH, 3), 256, 0, stream>>>(inp, inpu, inpd, tin, tinu, tind,
                                             ws_xn, ws_xbf, ws_ld);
  k_gemm<<<dim3(NSAMP / 64, 3), 256, 0, stream>>>(f0, f1, f2, ws_xbf, targets,
                                                  nf, ws_sum, ws_tlg);
  k_loss<<<1, 256, 0, stream>>>(ws_sum, ws_tlg, ws_ld, out);
  k_ema<<<3, 256, 0, stream>>>(targets, epoch, ws_xn, nf);
}
